// MultiHeadDiffAttention_6562710028759
// MI455X (gfx1250) — matrix operand order repaired and checked by reading, not compiled, not run on hardware
//
#include <hip/hip_runtime.h>
#include <hip/hip_bf16.h>
#include <math.h>

// ---------------------------------------------------------------------------
// Problem constants (B=2, T=2048, E=2048, H=16 from the reference)
// ---------------------------------------------------------------------------
#define T_SEQ   2048
#define EMB     2048
#define NROWS   4096      // B*T
#define QKV_N   6144      // 3*E
#define HD      64        // per-component head dim
#define LAMB0   0.2f      // LAMBDA_INIT

typedef __attribute__((ext_vector_type(16))) __bf16 v16bf;
typedef __attribute__((ext_vector_type(8)))  float  v8f;
typedef int v4i_ __attribute__((vector_size(16)));

union FragU { v16bf v; uint4 q[2]; unsigned short h[16]; };

#ifndef __has_builtin
#define __has_builtin(x) 0
#endif
#if __has_builtin(__builtin_amdgcn_global_load_async_to_lds_b128) && \
    __has_builtin(__builtin_amdgcn_s_wait_asynccnt)
#define ASYNC_LDS 1
#else
#define ASYNC_LDS 0
#endif

// Copy 8 bf16 (16B) global -> LDS. Async (ASYNCcnt-tracked, no VGPR bounce)
// when the gfx1250 async builtins exist; plain b128 load/store otherwise.
__device__ __forceinline__ void copy8_g2l(unsigned short* lds, const unsigned short* g) {
#if ASYNC_LDS
    __builtin_amdgcn_global_load_async_to_lds_b128(
        (__attribute__((address_space(1))) v4i_*)(g),
        (__attribute__((address_space(3))) v4i_*)(lds), 0, 0);
#else
    *(uint4*)lds = *(const uint4*)g;
#endif
}
__device__ __forceinline__ void async_join() {
#if ASYNC_LDS
    __builtin_amdgcn_s_wait_asynccnt(0);
#endif
}

__device__ __forceinline__ unsigned short f2bf(float f) {
    unsigned int u = __float_as_uint(f);
    u += 0x7fffu + ((u >> 16) & 1u);          // round-to-nearest-even
    return (unsigned short)(u >> 16);
}

// A-matrix 16x32 bf16 fragment from a row-major [16][ldk] tile (ISA 7.12.2).
__device__ __forceinline__ v16bf load_a_frag(const unsigned short* t, int ldk, int lane) {
    int row = lane & 15;
    int kb  = (lane & 16) ? 8 : 0;
    const unsigned short* p = t + row * ldk + kb;
    FragU f;
    f.q[0] = *(const uint4*)(p);
    f.q[1] = *(const uint4*)(p + 16);
    return f.v;
}

// B-matrix 32x16 bf16 fragment from an [N][ldk] (B^T row-major) tile.
__device__ __forceinline__ v16bf load_bt_frag(const unsigned short* t, int ldk, int lane) {
    int col = lane & 15;
    int kb  = (lane & 16) ? 8 : 0;
    const unsigned short* p = t + col * ldk + kb;
    FragU f;
    f.q[0] = *(const uint4*)(p);
    f.q[1] = *(const uint4*)(p + 16);
    return f.v;
}

__device__ __forceinline__ v8f wmma_bf16(v16bf a, v16bf b, v8f c) {
    return __builtin_amdgcn_wmma_f32_16x16x32_bf16(false, a, false, b,
                                                   (short)0, c, false, false);
}

// ---------------------------------------------------------------------------
// fp32 -> bf16 bulk convert
// ---------------------------------------------------------------------------
__global__ __launch_bounds__(256)
void f32_to_bf16_kernel(const float* __restrict__ in, unsigned short* __restrict__ out, long n) {
    long i = (long)blockIdx.x * blockDim.x + threadIdx.x;
    long stride = (long)gridDim.x * blockDim.x;
    for (; i < n; i += stride) out[i] = f2bf(in[i]);
}

// ---------------------------------------------------------------------------
// fp32 [K][N] -> bf16 [N][K] tiled transpose-convert (for weights: GEMM then
// consumes B as contiguous K-major rows, no LDS scatter in the hot loop).
// ---------------------------------------------------------------------------
__global__ __launch_bounds__(256)
void f32_to_bf16_T_kernel(const float* __restrict__ in, unsigned short* __restrict__ out,
                          int K, int N) {
    __shared__ unsigned short tile[32][33];
    const int kb = blockIdx.y * 32, nb = blockIdx.x * 32;
    const int tx = threadIdx.x & 31, ty = threadIdx.x >> 5;
    for (int i = 0; i < 32; i += 8)
        tile[ty + i][tx] = f2bf(in[(long)(kb + ty + i) * N + nb + tx]);
    __syncthreads();
    for (int i = 0; i < 32; i += 8)
        out[(long)(nb + ty + i) * K + kb + tx] = tile[tx][ty + i];
}

// ---------------------------------------------------------------------------
// bf16 transpose of the V panel of qkv: vtg[h*128+d][b*T+t] = qkv[row][4096+c]
// ---------------------------------------------------------------------------
__global__ __launch_bounds__(256)
void transpose_v_kernel(const unsigned short* __restrict__ qkv, unsigned short* __restrict__ vtg) {
    __shared__ unsigned short tile[32][33];
    const int rb = blockIdx.x * 32;      // token row
    const int cb = blockIdx.y * 32;      // v-dim
    const int tx = threadIdx.x & 31, ty = threadIdx.x >> 5;
    for (int i = 0; i < 32; i += 8)
        tile[ty + i][tx] = qkv[(long)(rb + ty + i) * QKV_N + 4096 + cb + tx];
    __syncthreads();
    for (int i = 0; i < 32; i += 8)
        vtg[(long)(cb + ty + i) * NROWS + rb + tx] = tile[tx][ty + i];
}

// ---------------------------------------------------------------------------
// lambda = exp(lq1.lk1) - exp(lq2.lk2) + 0.2   (single block)
// ---------------------------------------------------------------------------
__global__ __launch_bounds__(256)
void lambda_kernel(const float* lq1, const float* lk1,
                   const float* lq2, const float* lk2, float* out) {
    __shared__ float s1[256], s2[256];
    float a = 0.f, b = 0.f;
    for (int i = threadIdx.x; i < EMB; i += 256) {
        a += lq1[i] * lk1[i];
        b += lq2[i] * lk2[i];
    }
    s1[threadIdx.x] = a; s2[threadIdx.x] = b;
    __syncthreads();
    for (int st = 128; st > 0; st >>= 1) {
        if ((int)threadIdx.x < st) {
            s1[threadIdx.x] += s1[threadIdx.x + st];
            s2[threadIdx.x] += s2[threadIdx.x + st];
        }
        __syncthreads();
    }
    if (threadIdx.x == 0) out[0] = __expf(s1[0]) - __expf(s2[0]) + LAMB0;
}

// ---------------------------------------------------------------------------
// bf16 GEMM: C[M][N] = A[M][K] * BT[N][K]^T.  128x128 block tile, 256 threads
// (8 waves 2x4; each wave 64x32 = 4x2 WMMA tiles). Double-buffered LDS with
// async global->LDS fills; one barrier per K-step.
// ---------------------------------------------------------------------------
template<bool OUT_BF16>
__global__ __launch_bounds__(256)
void gemm_bf16_kernel(const unsigned short* __restrict__ A,   // [M][K]
                      const unsigned short* __restrict__ BT,  // [N][K]
                      void* __restrict__ C, int M, int N, int K)
{
    __shared__ unsigned short As[2][128][40];   // [m][k], ldk=40 (16B aligned)
    __shared__ unsigned short Bs[2][128][40];   // [n][k]

    const int t    = threadIdx.x;
    const int lane = t & 31;
    const int wave = t >> 5;
    const int wm   = wave >> 2;              // 0..1
    const int wn   = wave & 3;               // 0..3
    const int gm   = blockIdx.y * 128;
    const int gn   = blockIdx.x * 128;

    const int lr = t >> 1;                   // 0..127 tile row (m or n)
    const int lc = (t & 1) * 16;             // 0 or 16

    const unsigned short* gA = A  + (long)(gm + lr) * K + lc;
    const unsigned short* gB = BT + (long)(gn + lr) * K + lc;

    v8f acc[4][2];
    {
        v8f z = {};
        for (int i = 0; i < 4; ++i) for (int j = 0; j < 2; ++j) acc[i][j] = z;
    }

    auto load_tile = [&](int k0, int buf) {
        copy8_g2l(&As[buf][lr][lc],     gA + k0);
        copy8_g2l(&As[buf][lr][lc + 8], gA + k0 + 8);
        copy8_g2l(&Bs[buf][lr][lc],     gB + k0);
        copy8_g2l(&Bs[buf][lr][lc + 8], gB + k0 + 8);
    };

    load_tile(0, 0);
    async_join();
    __syncthreads();

    int cur = 0;
    for (int k0 = 0; k0 < K; k0 += 32) {
        const bool has_next = (k0 + 32 < K);
        if (has_next) load_tile(k0 + 32, cur ^ 1);

        v16bf af[4], bfg[2];
        for (int i = 0; i < 4; ++i) af[i]  = load_a_frag (&As[cur][wm * 64 + i * 16][0], 40, lane);
        for (int j = 0; j < 2; ++j) bfg[j] = load_bt_frag(&Bs[cur][wn * 32 + j * 16][0], 40, lane);
        for (int i = 0; i < 4; ++i)
            for (int j = 0; j < 2; ++j)
                acc[i][j] = wmma_bf16(af[i], bfg[j], acc[i][j]);

        if (has_next) {
            async_join();
            __syncthreads();
            cur ^= 1;
        }
    }

    // C/D layout: element r of lane = (row r + 8*(lane>=16), col lane&15)
    const int rowoff = (lane >> 4) << 3;
    const int col    = lane & 15;
    for (int i = 0; i < 4; ++i)
        for (int j = 0; j < 2; ++j)
            for (int r = 0; r < 8; ++r) {
                long row = gm + wm * 64 + i * 16 + r + rowoff;
                long cc  = gn + wn * 32 + j * 16 + col;
                float v  = acc[i][j][r];
                if (OUT_BF16) ((unsigned short*)C)[row * N + cc] = f2bf(v);
                else          ((float*)C)[row * N + cc] = v;
            }
}

// ---------------------------------------------------------------------------
// Differential flash attention.
// Grid: (T/64, H, B).  Block: 128 threads = 4 waves, each wave = 16 q-rows.
// Double-buffered 32-key tiles (K1, K2 key-major; V^T dim-major), online
// softmax of (S1 - lam*S2)*scale, P@V via WMMA.
// ---------------------------------------------------------------------------
__global__ __launch_bounds__(128)
void flash_diff_attn(const unsigned short* __restrict__ qkv,
                     const unsigned short* __restrict__ vtg,  // [E][NROWS]
                     const float* __restrict__ lam_p,
                     unsigned short* __restrict__ ao)
{
    __shared__ unsigned short k1s[2][32][72];    // [key][hd]  (== B^T for QK^T)
    __shared__ unsigned short k2s[2][32][72];
    __shared__ unsigned short vt [2][128][40];   // [dim][key] (== B^T for P@V)
    __shared__ unsigned short pl [4][16][40];    // per-wave staged P (A layout)

    const int t    = threadIdx.x;
    const int lane = t & 31;
    const int wave = t >> 5;
    const int b    = blockIdx.z;
    const int h    = blockIdx.y;

    const float lam   = *lam_p;
    const float scale = 0.125f;                  // 64^-0.5

    const int q1off = 0    + h * HD;
    const int k1off = 1024 + h * HD;
    const int q2off = 2048 + h * HD;
    const int k2off = 3072 + h * HD;

    const long rowbase = (long)b * T_SEQ;
    const int  qrow0   = blockIdx.x * 64 + wave * 16;

    // Q fragments (A layout) loaded straight from global, kept in registers.
    v16bf qa1[2], qa2[2];
    {
        const unsigned short* qp = qkv + (rowbase + qrow0) * QKV_N;
        for (int kk = 0; kk < 2; ++kk) {
            qa1[kk] = load_a_frag(qp + q1off + kk * 32, QKV_N, lane);
            qa2[kk] = load_a_frag(qp + q2off + kk * 32, QKV_N, lane);
        }
    }

    auto load_kv = [&](int kt, int buf) {
        const long krow = rowbase + (long)kt * 32;
        const int  rr = t >> 2;                  // 0..31 key
        const int  cc = (t & 3) * 16;            // 0,16,32,48 hd
        const unsigned short* g1 = qkv + (krow + rr) * QKV_N + k1off + cc;
        copy8_g2l(&k1s[buf][rr][cc],     g1);
        copy8_g2l(&k1s[buf][rr][cc + 8], g1 + 8);
        const unsigned short* g2 = qkv + (krow + rr) * QKV_N + k2off + cc;
        copy8_g2l(&k2s[buf][rr][cc],     g2);
        copy8_g2l(&k2s[buf][rr][cc + 8], g2 + 8);
        const int d = t;                         // 0..127 v-dim
        const unsigned short* gv = vtg + (long)(h * 128 + d) * NROWS + rowbase + (long)kt * 32;
        copy8_g2l(&vt[buf][d][0],  gv);
        copy8_g2l(&vt[buf][d][8],  gv + 8);
        copy8_g2l(&vt[buf][d][16], gv + 16);
        copy8_g2l(&vt[buf][d][24], gv + 24);
    };

    float mrun[8], lrun[8];
    v8f   acc[8];
    {
        v8f z = {};
        for (int r = 0; r < 8; ++r) { mrun[r] = -1e30f; lrun[r] = 0.f; }
        for (int c = 0; c < 8; ++c) acc[c] = z;
    }

    load_kv(0, 0);
    async_join();
    __syncthreads();

    int cur = 0;
    for (int kt = 0; kt < T_SEQ / 32; ++kt) {
        const bool has_next = (kt + 1 < T_SEQ / 32);
        if (has_next) load_kv(kt + 1, cur ^ 1);

        // S1/S2 for two 16-key chunks; fuse differential score.
        v8f p0, p1;
        {
            v8f z = {};
            v8f s1 = z, s2 = z;
            s1 = wmma_bf16(qa1[0], load_bt_frag(&k1s[cur][0][0],  72, lane), s1);
            s1 = wmma_bf16(qa1[1], load_bt_frag(&k1s[cur][0][32], 72, lane), s1);
            s2 = wmma_bf16(qa2[0], load_bt_frag(&k2s[cur][0][0],  72, lane), s2);
            s2 = wmma_bf16(qa2[1], load_bt_frag(&k2s[cur][0][32], 72, lane), s2);
            p0 = (s1 - lam * s2) * scale;
            s1 = z; s2 = z;
            s1 = wmma_bf16(qa1[0], load_bt_frag(&k1s[cur][16][0],  72, lane), s1);
            s1 = wmma_bf16(qa1[1], load_bt_frag(&k1s[cur][16][32], 72, lane), s1);
            s2 = wmma_bf16(qa2[0], load_bt_frag(&k2s[cur][16][0],  72, lane), s2);
            s2 = wmma_bf16(qa2[1], load_bt_frag(&k2s[cur][16][32], 72, lane), s2);
            p1 = (s1 - lam * s2) * scale;
        }

        // Online softmax. Rows live in 16-lane halves; xor masks 1,2,4,8
        // reduce within a half on wave32.
        float corr[8];
        for (int r = 0; r < 8; ++r) {
            float mx = fmaxf(p0[r], p1[r]);
            for (int s = 8; s >= 1; s >>= 1)
                mx = fmaxf(mx, __shfl_xor(mx, s, 32));
            float nm = fmaxf(mrun[r], mx);
            corr[r]  = __expf(mrun[r] - nm);
            mrun[r]  = nm;
            float e0 = __expf(p0[r] - nm);
            float e1 = __expf(p1[r] - nm);
            p0[r] = e0; p1[r] = e1;
            float rs = e0 + e1;
            for (int s = 8; s >= 1; s >>= 1)
                rs += __shfl_xor(rs, s, 32);
            lrun[r] = lrun[r] * corr[r] + rs;
        }

        // Stage P (C layout -> per-wave LDS -> A layout). Same-wave DS ops are
        // in-order on CDNA5; the asm barrier only pins compiler ordering.
        {
            const int rowoff = (lane >> 4) << 3;
            const int col    = lane & 15;
            for (int r = 0; r < 8; ++r) {
                pl[wave][r + rowoff][col]      = f2bf(p0[r]);
                pl[wave][r + rowoff][16 + col] = f2bf(p1[r]);
            }
        }
        __asm__ volatile("" ::: "memory");
        v16bf pfrag = load_a_frag(&pl[wave][0][0], 40, lane);
        for (int c = 0; c < 8; ++c) {
            v8f a = acc[c];
            for (int r = 0; r < 8; ++r) a[r] *= corr[r];
            acc[c] = wmma_bf16(pfrag, load_bt_frag(&vt[cur][c * 16][0], 40, lane), a);
        }

        if (has_next) {
            async_join();
            __syncthreads();
            cur ^= 1;
        }
    }

    // Epilogue: out = 0.8 * acc / l, bf16 into ao[row][h*128 + dim]
    const int rowoff = (lane >> 4) << 3;
    const int col    = lane & 15;
    float inv[8];
    for (int r = 0; r < 8; ++r) inv[r] = 0.8f / lrun[r];
    for (int c = 0; c < 8; ++c)
        for (int r = 0; r < 8; ++r) {
            long row = rowbase + qrow0 + r + rowoff;
            ao[row * EMB + h * 128 + c * 16 + col] = f2bf(acc[c][r] * inv[r]);
        }
}

// ---------------------------------------------------------------------------
// Launch
// ---------------------------------------------------------------------------
extern "C" void kernel_launch(void* const* d_in, const int* in_sizes, int n_in,
                              void* d_out, int out_size, void* d_ws, size_t ws_size,
                              hipStream_t stream) {
    (void)in_sizes; (void)n_in; (void)out_size; (void)ws_size;
    const float* x      = (const float*)d_in[0];
    const float* W_attn = (const float*)d_in[1];
    const float* W_proj = (const float*)d_in[2];
    const float* lq1    = (const float*)d_in[3];
    const float* lk1    = (const float*)d_in[4];
    const float* lq2    = (const float*)d_in[5];
    const float* lk2    = (const float*)d_in[6];

    char* ws = (char*)d_ws;
    unsigned short* xb  = (unsigned short*)ws;  ws += (long)NROWS * EMB   * 2;
    unsigned short* waT = (unsigned short*)ws;  ws += (long)EMB   * QKV_N * 2;  // [QKV_N][EMB]
    unsigned short* wpT = (unsigned short*)ws;  ws += (long)EMB   * EMB   * 2;  // [EMB][EMB]
    unsigned short* qkv = (unsigned short*)ws;  ws += (long)NROWS * QKV_N * 2;
    unsigned short* vtg = (unsigned short*)ws;  ws += (long)EMB   * NROWS * 2;  // [E][NROWS]
    unsigned short* ao  = (unsigned short*)ws;  ws += (long)NROWS * EMB   * 2;
    float*          lam = (float*)ws;

    f32_to_bf16_kernel<<<2048, 256, 0, stream>>>(x, xb, (long)NROWS * EMB);
    f32_to_bf16_T_kernel<<<dim3(QKV_N / 32, EMB / 32), 256, 0, stream>>>(W_attn, waT, EMB, QKV_N);
    f32_to_bf16_T_kernel<<<dim3(EMB   / 32, EMB / 32), 256, 0, stream>>>(W_proj, wpT, EMB, EMB);
    lambda_kernel<<<1, 256, 0, stream>>>(lq1, lk1, lq2, lk2, lam);

    // qkv = xb @ waT^T : [4096 x 2048] * [2048 x 6144]
    gemm_bf16_kernel<true ><<<dim3(QKV_N / 128, NROWS / 128), 256, 0, stream>>>(
        xb, waT, qkv, NROWS, QKV_N, EMB);

    // V panel -> V^T for B-fragment-friendly attention loads
    transpose_v_kernel<<<dim3(NROWS / 32, EMB / 32), 256, 0, stream>>>(qkv, vtg);

    // differential flash attention -> ao (bf16, includes 0.8 scale)
    flash_diff_attn<<<dim3(T_SEQ / 64, 16, 2), 128, 0, stream>>>(qkv, vtg, lam, ao);

    // d_out = ao @ wpT^T : [4096 x 2048] * [2048 x 2048], fp32 out
    gemm_bf16_kernel<false><<<dim3(EMB / 128, NROWS / 128), 256, 0, stream>>>(
        ao, wpT, (float*)d_out, NROWS, EMB, EMB);
}
